// BigBirdEncoder_69020124446908
// MI455X (gfx1250) — compile-verified, matrix operand order checked by reading, not run on hardware
//
#include <hip/hip_runtime.h>

// ---------------------------------------------------------------------------
// BigBird encoder forward, MI455X (gfx1250), bf16 WMMA everywhere.
// B=4 S=2048 D=512 H=8 DH=64 M=2048 L=6 BS=64 NB=32 Kc=8
// ---------------------------------------------------------------------------

typedef __bf16 v16bf __attribute__((ext_vector_type(16)));
typedef float  v8f   __attribute__((ext_vector_type(8)));

#define NLAYER 6
#define NBLK   32
#define KC     8

__device__ __forceinline__ __bf16 f2bf(float f) {
    union { float f; unsigned u; } in; in.f = f;
    unsigned r = (in.u + 0x7fffu + ((in.u >> 16) & 1u)) >> 16;
    union { unsigned short u; __bf16 b; } out; out.u = (unsigned short)r;
    return out.b;
}

// ---------------------------------------------------------------------------
// fp32 -> bf16 copy (weights)
// ---------------------------------------------------------------------------
__global__ __launch_bounds__(256) void cvt_kernel(const float* __restrict__ s,
                                                  __bf16* __restrict__ d, int n) {
    int i = blockIdx.x * 256 + threadIdx.x;
    if (i < n) d[i] = f2bf(s[i]);
}

// ---------------------------------------------------------------------------
// token embedding + sinusoidal positional encoding
// ---------------------------------------------------------------------------
__global__ __launch_bounds__(256) void embed_kernel(const int* __restrict__ tokens,
                                                    const float* __restrict__ emb,
                                                    float* __restrict__ x) {
    int bs = blockIdx.x;                 // 0..8191  (b*2048 + s)
    int s  = bs & 2047;
    int tok = tokens[bs];
    const float* e = emb + (size_t)tok * 512;
    float* xr = x + (size_t)bs * 512;
    for (int c = threadIdx.x; c < 512; c += 256) {
        int i = c >> 1;
        float dv  = __expf((-9.210340371976184f / 512.f) * (float)(2 * i));
        float ang = (float)s * dv;
        float pe  = (c & 1) ? __cosf(ang) : __sinf(ang);
        xr[c] = e[c] + pe;
    }
}

// ---------------------------------------------------------------------------
// LayerNorm over D=512, one wave per row.  BF=true -> bf16 out, else fp32 out.
// ---------------------------------------------------------------------------
template <bool BF>
__global__ __launch_bounds__(256) void ln_kernel(const float* __restrict__ x,
                                                 const float* __restrict__ sc,
                                                 const float* __restrict__ bi,
                                                 __bf16* __restrict__ obf,
                                                 float* __restrict__ of32) {
    int row  = blockIdx.x * 8 + (threadIdx.x >> 5);
    int lane = threadIdx.x & 31;
    const float* xr = x + (size_t)row * 512;
    float v[16];
#pragma unroll
    for (int q = 0; q < 4; ++q)
        *(float4*)&v[q * 4] = *(const float4*)(xr + lane * 16 + q * 4);
    float sum = 0.f;
#pragma unroll
    for (int i = 0; i < 16; ++i) sum += v[i];
#pragma unroll
    for (int d = 1; d < 32; d <<= 1) sum += __shfl_xor(sum, d, 32);
    float mean = sum * (1.f / 512.f);
    float var = 0.f;
#pragma unroll
    for (int i = 0; i < 16; ++i) { float t = v[i] - mean; var += t * t; }
#pragma unroll
    for (int d = 1; d < 32; d <<= 1) var += __shfl_xor(var, d, 32);
    float rstd = rsqrtf(var * (1.f / 512.f) + 1e-6f);
#pragma unroll
    for (int i = 0; i < 16; ++i) {
        int col = lane * 16 + i;
        float y = (v[i] - mean) * rstd * sc[col] + bi[col];
        if (BF) obf[(size_t)row * 512 + col] = f2bf(y);
        else    of32[(size_t)row * 512 + col] = y;
    }
}

// ---------------------------------------------------------------------------
// Tiled bf16 WMMA GEMM:  C[M,N] = A[M,K] * B[K,N]   (A,B bf16 row-major)
// Block tile 128x64x32, 8 waves, each wave a 32x32 tile (4 wmma accumulators).
// ---------------------------------------------------------------------------
#define EPI_Q      0  // scatter to q  [B,H,NB,64,64]
#define EPI_K      1  // scatter to k  [B,H,NB,64,64]
#define EPI_VT     2  // scatter to vT [B,H,NB,64(dh),64(q)]
#define EPI_ADD    3  // Cadd[row*N+col] += acc
#define EPI_GELU_B 4  // Obf = bf16(gelu(acc + bias[col]))
#define EPI_ADD_B  5  // Cadd[row*N+col] += acc + bias[col]

template <int EPI>
__global__ __launch_bounds__(256) void gemm_bf16(const __bf16* __restrict__ A,
                                                 const __bf16* __restrict__ Bm,
                                                 int M, int N, int K,
                                                 float* __restrict__ Cadd,
                                                 __bf16* __restrict__ Obf,
                                                 const float* __restrict__ bias) {
    __shared__ __bf16 As[128][40];   // row stride 80B: 16B aligned, conflict-free
    __shared__ __bf16 Bs[64][40];    // stored transposed: Bs[n][k]

    const int t = threadIdx.x;
    const int wave = t >> 5, lane = t & 31;
    const int half = lane >> 4, l16 = lane & 15;
    const int wm = wave >> 1, wn = wave & 1;
    const int row0 = blockIdx.y * 128;
    const int n0   = blockIdx.x * 64;

    const int arow = t >> 1, acol0 = (t & 1) * 16;
    const int brow = t >> 3, bcol0 = (t & 7) * 8;

    v8f acc[2][2] = {};
    const int nk = K >> 5;
    for (int kt = 0; kt < nk; ++kt) {
        __syncthreads();
        // ---- stage A (128x32) ----
        const uint4* ga = (const uint4*)(A + (size_t)(row0 + arow) * K + kt * 32 + acol0);
        uint4 a0 = ga[0], a1 = ga[1];
        *(uint4*)&As[arow][acol0]     = a0;
        *(uint4*)&As[arow][acol0 + 8] = a1;
        // ---- stage B (32x64) transposed ----
        union { uint4 v; __bf16 h[8]; } ub;
        ub.v = *(const uint4*)(Bm + (size_t)(kt * 32 + brow) * N + n0 + bcol0);
#pragma unroll
        for (int j = 0; j < 8; ++j) Bs[bcol0 + j][brow] = ub.h[j];
        if (kt + 1 < nk) {
            __builtin_prefetch(A + (size_t)(row0 + arow) * K + (kt + 1) * 32 + acol0, 0, 3);
            __builtin_prefetch(Bm + (size_t)((kt + 1) * 32 + brow) * N + n0 + bcol0, 0, 3);
        }
        __syncthreads();
        // ---- fragments + wmma ----
        v16bf af[2], bf[2];
#pragma unroll
        for (int mt = 0; mt < 2; ++mt)
            af[mt] = *(const v16bf*)&As[wm * 32 + mt * 16 + l16][half * 16];
#pragma unroll
        for (int nt = 0; nt < 2; ++nt)
            bf[nt] = *(const v16bf*)&Bs[wn * 32 + nt * 16 + l16][half * 16];
#pragma unroll
        for (int mt = 0; mt < 2; ++mt)
#pragma unroll
            for (int nt = 0; nt < 2; ++nt)
                acc[mt][nt] = __builtin_amdgcn_wmma_f32_16x16x32_bf16(
                    false, af[mt], false, bf[nt], (short)0, acc[mt][nt], false, false);
    }

    // ---- epilogue ----
#pragma unroll
    for (int mt = 0; mt < 2; ++mt)
#pragma unroll
        for (int nt = 0; nt < 2; ++nt)
#pragma unroll
            for (int r = 0; r < 8; ++r) {
                int row = row0 + wm * 32 + mt * 16 + half * 8 + r;
                int col = n0 + wn * 32 + nt * 16 + l16;
                float v = acc[mt][nt][r];
                if (EPI == EPI_Q || EPI == EPI_K || EPI == EPI_VT) {
                    int b = row >> 11, s = row & 2047;
                    int n = s >> 6, qi = s & 63;
                    int hh = col >> 6, dh = col & 63;
                    size_t base = (((size_t)b * 8 + hh) * 32 + n) * 4096;
                    size_t idx = (EPI == EPI_VT) ? base + (size_t)dh * 64 + qi
                                                 : base + (size_t)qi * 64 + dh;
                    Obf[idx] = f2bf(v);
                } else if (EPI == EPI_ADD) {
                    Cadd[(size_t)row * N + col] += v;
                } else if (EPI == EPI_GELU_B) {
                    float u = v + bias[col];
                    float g = 0.5f * u *
                              (1.f + tanhf(0.7978845608028654f * (u + 0.044715f * u * u * u)));
                    Obf[(size_t)row * N + col] = f2bf(g);
                } else {  // EPI_ADD_B
                    Cadd[(size_t)row * N + col] += v + bias[col];
                }
            }
}

// ---------------------------------------------------------------------------
// BigBird block-sparse attention (flash-style online softmax).
// One workgroup per (b, n, h); 4 waves x 16 query rows.
// q/k : [B,H,NB,64,64] bf16 ; vT : [B,H,NB,64(dh),64(key)] bf16
// ---------------------------------------------------------------------------
__global__ __launch_bounds__(128) void attn_kernel(const __bf16* __restrict__ qbuf,
                                                   const __bf16* __restrict__ kbuf,
                                                   const __bf16* __restrict__ vtbuf,
                                                   const int* __restrict__ tokens,
                                                   const int* __restrict__ pidx,
                                                   const unsigned char* __restrict__ pmsk,
                                                   __bf16* __restrict__ aout) {
    const int bid = blockIdx.x;                  // b*256 + n*8 + h
    const int h = bid & 7, n = (bid >> 3) & 31, b = bid >> 8;
    const int t = threadIdx.x, w = t >> 5, lane = t & 31;
    const int half = lane >> 4, l16 = lane & 15;

    __shared__ __bf16 pT[4][16][72];             // per-wave P tile, 16x64, padded

    const __bf16* qb = qbuf + ((((size_t)b * 8 + h) * 32 + n) << 12);
    v16bf qf[2];
    qf[0] = *(const v16bf*)(qb + (size_t)(w * 16 + l16) * 64 + half * 16);
    qf[1] = *(const v16bf*)(qb + (size_t)(w * 16 + l16) * 64 + 32 + half * 16);

    v8f oacc[4] = {};
    float rmax[8], rsum[8];
#pragma unroll
    for (int r = 0; r < 8; ++r) { rmax[r] = -1e30f; rsum[r] = 0.f; }

    const int* tok = tokens + b * 2048;

    for (int kc = 0; kc < KC; ++kc) {
        if (!pmsk[n * KC + kc]) continue;        // block-uniform branch
        int g = pidx[n * KC + kc];
        const __bf16* kb = kbuf + ((((size_t)b * 8 + h) * 32 + g) << 12);
        const __bf16* vb = vtbuf + ((((size_t)b * 8 + h) * 32 + g) << 12);

        // ---- S = Q * K^T  (16x64 per wave) ----
        v8f sA[4] = {};
#pragma unroll
        for (int ks = 0; ks < 2; ++ks)
#pragma unroll
            for (int nt = 0; nt < 4; ++nt) {
                v16bf bf = *(const v16bf*)(kb + (size_t)(nt * 16 + l16) * 64 + ks * 32 + half * 16);
                sA[nt] = __builtin_amdgcn_wmma_f32_16x16x32_bf16(
                    false, qf[ks], false, bf, (short)0, sA[nt], false, false);
            }

        // ---- scale + key-padding mask ----
        float sv[4][8];
#pragma unroll
        for (int nt = 0; nt < 4; ++nt) {
            bool ok = tok[g * 64 + nt * 16 + l16] > 0;
#pragma unroll
            for (int r = 0; r < 8; ++r)
                sv[nt][r] = ok ? sA[nt][r] * 0.125f : -1e9f;
        }

        // ---- online softmax update ----
#pragma unroll
        for (int r = 0; r < 8; ++r) {
            float lm = fmaxf(fmaxf(sv[0][r], sv[1][r]), fmaxf(sv[2][r], sv[3][r]));
#pragma unroll
            for (int d = 1; d < 16; d <<= 1) lm = fmaxf(lm, __shfl_xor(lm, d, 16));
            float mnew = fmaxf(rmax[r], lm);
            float corr = __expf(rmax[r] - mnew);
            float ls = 0.f;
#pragma unroll
            for (int nt = 0; nt < 4; ++nt) {
                float p = __expf(sv[nt][r] - mnew);
                sv[nt][r] = p;
                ls += p;
            }
#pragma unroll
            for (int d = 1; d < 16; d <<= 1) ls += __shfl_xor(ls, d, 16);
            rsum[r] = rsum[r] * corr + ls;
            rmax[r] = mnew;
#pragma unroll
            for (int nt = 0; nt < 4; ++nt) oacc[nt][r] *= corr;
        }

        // ---- P (D-layout) -> LDS -> A-fragment layout ----
#pragma unroll
        for (int nt = 0; nt < 4; ++nt)
#pragma unroll
            for (int r = 0; r < 8; ++r)
                pT[w][half * 8 + r][nt * 16 + l16] = f2bf(sv[nt][r]);
        __syncthreads();

        // ---- O += P * V ----
#pragma unroll
        for (int ks = 0; ks < 2; ++ks) {
            v16bf pf = *(const v16bf*)&pT[w][l16][ks * 32 + half * 16];
#pragma unroll
            for (int nt = 0; nt < 4; ++nt) {
                v16bf vf = *(const v16bf*)(vb + (size_t)(nt * 16 + l16) * 64 + ks * 32 + half * 16);
                oacc[nt] = __builtin_amdgcn_wmma_f32_16x16x32_bf16(
                    false, pf, false, vf, (short)0, oacc[nt], false, false);
            }
        }
        __syncthreads();
    }

    // ---- normalize + write [8192, 512] bf16 ----
    int srow0 = b * 2048 + n * 64 + w * 16 + half * 8;
#pragma unroll
    for (int r = 0; r < 8; ++r) {
        float inv = rsum[r] > 0.f ? 1.f / rsum[r] : 0.f;
#pragma unroll
        for (int nt = 0; nt < 4; ++nt) {
            int col = h * 64 + nt * 16 + l16;
            aout[(size_t)(srow0 + r) * 512 + col] = f2bf(oacc[nt][r] * inv);
        }
    }
}

// ---------------------------------------------------------------------------
// numpy-RandomState-compatible BigBird block plan, computed on device
// (single thread; graph-capture safe, no host->device copies).
// ---------------------------------------------------------------------------
struct MT19937 {
    unsigned mt[624];
    int mti;
    __device__ void seed(unsigned s) {
        mt[0] = s;
        for (int i = 1; i < 624; ++i)
            mt[i] = 1812433253u * (mt[i - 1] ^ (mt[i - 1] >> 30)) + (unsigned)i;
        mti = 624;
    }
    __device__ unsigned next() {
        if (mti >= 624) {
            for (int i = 0; i < 624; ++i) {
                unsigned y = (mt[i] & 0x80000000u) | (mt[(i + 1) % 624] & 0x7fffffffu);
                mt[i] = mt[(i + 397) % 624] ^ (y >> 1) ^ ((y & 1u) ? 2567483615u : 0u);
            }
            mti = 0;
        }
        unsigned y = mt[mti++];
        y ^= y >> 11; y ^= (y << 7) & 2636928640u; y ^= (y << 15) & 4022730752u; y ^= y >> 18;
        return y;
    }
    __device__ unsigned interval(unsigned mx) {  // uniform in [0, mx]
        if (mx == 0) return 0;
        unsigned m = mx;
        m |= m >> 1; m |= m >> 2; m |= m >> 4; m |= m >> 8; m |= m >> 16;
        unsigned v;
        do { v = next() & m; } while (v > mx);
        return v;
    }
};

__global__ void plan_kernel(int* __restrict__ idx, unsigned char* __restrict__ msk) {
    if (threadIdx.x || blockIdx.x) return;
    MT19937 rng;
    for (int l = 0; l < NLAYER; ++l) {
        rng.seed((unsigned)l);
        for (int i = 0; i < NBLK; ++i) {
            bool sel[NBLK];
            for (int j = 0; j < NBLK; ++j) sel[j] = false;
            sel[0] = sel[NBLK - 1] = true;
            for (int j = i - 1; j <= i + 1; ++j)
                if (j >= 0 && j < NBLK) sel[j] = true;
            int rest[NBLK], nr = 0;
            for (int j = 0; j < NBLK; ++j)
                if (!sel[j]) rest[nr++] = j;
            int perm[NBLK];
            for (int j = 0; j < nr; ++j) perm[j] = j;
            for (int ii = nr - 1; ii >= 1; --ii) {  // numpy Fisher-Yates
                unsigned jj = rng.interval((unsigned)ii);
                int tmp = perm[ii]; perm[ii] = perm[jj]; perm[jj] = tmp;
            }
            for (int j = 0; j < 3; ++j) sel[rest[perm[j]]] = true;
            int cnt = 0;
            for (int j = 0; j < NBLK; ++j)
                if (sel[j]) { idx[(l * NBLK + i) * KC + cnt] = j;
                              msk[(l * NBLK + i) * KC + cnt] = 1; ++cnt; }
            for (; cnt < KC; ++cnt) { idx[(l * NBLK + i) * KC + cnt] = 0;
                                      msk[(l * NBLK + i) * KC + cnt] = 0; }
        }
    }
}

// ---------------------------------------------------------------------------
// host driver
// ---------------------------------------------------------------------------
extern "C" void kernel_launch(void* const* d_in, const int* in_sizes, int n_in,
                              void* d_out, int out_size, void* d_ws, size_t ws_size,
                              hipStream_t stream) {
    const int*   tokens = (const int*)d_in[0];
    const float* embed  = (const float*)d_in[1];
    const float* ln1_s  = (const float*)d_in[2];
    const float* ln1_b  = (const float*)d_in[3];
    const float* wq     = (const float*)d_in[4];
    const float* wk     = (const float*)d_in[5];
    const float* wv     = (const float*)d_in[6];
    const float* wo     = (const float*)d_in[7];
    const float* ln2_s  = (const float*)d_in[8];
    const float* ln2_b  = (const float*)d_in[9];
    const float* w1     = (const float*)d_in[10];
    const float* b1     = (const float*)d_in[11];
    const float* w2     = (const float*)d_in[12];
    const float* b2     = (const float*)d_in[13];
    const float* lnf_s  = (const float*)d_in[14];
    const float* lnf_b  = (const float*)d_in[15];

    char* ws = (char*)d_ws;
    size_t off = 0;
    auto alloc = [&](size_t bytes) -> void* {
        void* p = ws + off;
        off += (bytes + 255) & ~(size_t)255;
        return p;
    };
    float*  x    = (float*)alloc((size_t)8192 * 512 * 4);     // residual stream
    __bf16* lnb  = (__bf16*)alloc((size_t)8192 * 512 * 2);    // LN output (bf16)
    __bf16* qb   = (__bf16*)alloc((size_t)4194304 * 2);       // [B,H,NB,64,64]
    __bf16* kb   = (__bf16*)alloc((size_t)4194304 * 2);
    __bf16* vtb  = (__bf16*)alloc((size_t)4194304 * 2);       // V transposed
    __bf16* ao   = (__bf16*)alloc((size_t)8192 * 512 * 2);    // attn out (bf16)
    __bf16* hb   = (__bf16*)alloc((size_t)8192 * 2048 * 2);   // FFN hidden (bf16)
    __bf16* wbq  = (__bf16*)alloc((size_t)262144 * 2);
    __bf16* wbk  = (__bf16*)alloc((size_t)262144 * 2);
    __bf16* wbv  = (__bf16*)alloc((size_t)262144 * 2);
    __bf16* wbo  = (__bf16*)alloc((size_t)262144 * 2);
    __bf16* wb1  = (__bf16*)alloc((size_t)1048576 * 2);
    __bf16* wb2  = (__bf16*)alloc((size_t)1048576 * 2);
    int*    pidx = (int*)alloc((size_t)NLAYER * NBLK * KC * 4);
    unsigned char* pmsk = (unsigned char*)alloc((size_t)NLAYER * NBLK * KC);
    (void)ws_size; (void)n_in; (void)in_sizes; (void)out_size;

    plan_kernel<<<1, 1, 0, stream>>>(pidx, pmsk);
    embed_kernel<<<8192, 256, 0, stream>>>(tokens, embed, x);

    const dim3 g512(8, 64);    // N=512 : 8 x 64 tiles of 64x128
    const dim3 g2048(32, 64);  // N=2048

    for (int l = 0; l < NLAYER; ++l) {
        // LN1 -> bf16
        ln_kernel<true><<<1024, 256, 0, stream>>>(x, ln1_s + l * 512, ln1_b + l * 512,
                                                  lnb, nullptr);
        // convert this layer's weights to bf16
        cvt_kernel<<<1024, 256, 0, stream>>>(wq + (size_t)l * 262144, wbq, 262144);
        cvt_kernel<<<1024, 256, 0, stream>>>(wk + (size_t)l * 262144, wbk, 262144);
        cvt_kernel<<<1024, 256, 0, stream>>>(wv + (size_t)l * 262144, wbv, 262144);
        cvt_kernel<<<1024, 256, 0, stream>>>(wo + (size_t)l * 262144, wbo, 262144);
        cvt_kernel<<<4096, 256, 0, stream>>>(w1 + (size_t)l * 1048576, wb1, 1048576);
        cvt_kernel<<<4096, 256, 0, stream>>>(w2 + (size_t)l * 1048576, wb2, 1048576);
        // QKV projections (WMMA)
        gemm_bf16<EPI_Q><<<g512, 256, 0, stream>>>(lnb, wbq, 8192, 512, 512,
                                                   nullptr, qb, nullptr);
        gemm_bf16<EPI_K><<<g512, 256, 0, stream>>>(lnb, wbk, 8192, 512, 512,
                                                   nullptr, kb, nullptr);
        gemm_bf16<EPI_VT><<<g512, 256, 0, stream>>>(lnb, wbv, 8192, 512, 512,
                                                    nullptr, vtb, nullptr);
        // block-sparse attention (WMMA QK^T + PV, online softmax)
        attn_kernel<<<1024, 128, 0, stream>>>(qb, kb, vtb, tokens,
                                              pidx + l * NBLK * KC,
                                              pmsk + l * NBLK * KC, ao);
        // output projection, residual add into x
        gemm_bf16<EPI_ADD><<<g512, 256, 0, stream>>>(ao, wbo, 8192, 512, 512,
                                                     x, nullptr, nullptr);
        // LN2 -> bf16
        ln_kernel<true><<<1024, 256, 0, stream>>>(x, ln2_s + l * 512, ln2_b + l * 512,
                                                  lnb, nullptr);
        // FFN
        gemm_bf16<EPI_GELU_B><<<g2048, 256, 0, stream>>>(lnb, wb1, 8192, 2048, 512,
                                                         nullptr, hb, b1 + (size_t)l * 2048);
        gemm_bf16<EPI_ADD_B><<<g512, 256, 0, stream>>>(hb, wb2, 8192, 512, 2048,
                                                       x, nullptr, b2 + (size_t)l * 512);
    }
    // final LN -> fp32 output
    ln_kernel<false><<<1024, 256, 0, stream>>>(x, lnf_s, lnf_b, nullptr, (float*)d_out);
}
